// KernelExpFamily_45277545234446
// MI455X (gfx1250) — matrix operand order repaired and checked by reading, not compiled, not run on hardware
//
#include <hip/hip_runtime.h>

typedef __attribute__((ext_vector_type(16))) _Float16 v16h;
typedef __attribute__((ext_vector_type(8)))  _Float16 v8h;
typedef __attribute__((ext_vector_type(8)))  float    v8f;

#define FEAT    256
#define BATCH   8192
#define DB      16384
#define DB_PER  512          // DB rows per workgroup (grid.y = DB / DB_PER = 32)
#define LDS_STRIDE 264       // 256 + 8 halfs pad -> conflict-free ds_load_b128
#define LOG2E   1.4426950408889634f

#if defined(__has_builtin) && __has_builtin(__builtin_amdgcn_exp2f)
#define FAST_EXP2(x) __builtin_amdgcn_exp2f(x)   // bare v_exp_f32 (arg range ~ +-0.5, safe)
#else
#define FAST_EXP2(x) exp2f(x)
#endif

__global__ void zero_out(float* p, int n) {
  int i = blockIdx.x * blockDim.x + threadIdx.x;
  if (i < n) p[i] = 0.0f;
}

// One block per row, 256 threads: f32 -> f16 convert + fused RBF prefactor.
// mode 0 (x side):  outv[row] = exp(-gamma * ||row||^2)
// mode 1 (X side):  outv[row] = alpha[row] * exp(-gamma * ||row||^2)
__global__ void prep_rows(const float* __restrict__ src, _Float16* __restrict__ dst,
                          float* __restrict__ outv, const float* __restrict__ alpha,
                          const float* __restrict__ gammap, int mode) {
  __shared__ float red[8];
  const int row = blockIdx.x;
  const int tid = threadIdx.x;
  float v = src[(size_t)row * FEAT + tid];
  dst[(size_t)row * FEAT + tid] = (_Float16)v;
  float s = v * v;
  s += __shfl_xor(s, 16, 32);
  s += __shfl_xor(s, 8, 32);
  s += __shfl_xor(s, 4, 32);
  s += __shfl_xor(s, 2, 32);
  s += __shfl_xor(s, 1, 32);
  if ((tid & 31) == 0) red[tid >> 5] = s;
  __syncthreads();
  if (tid == 0) {
    float t = 0.f;
    #pragma unroll
    for (int i = 0; i < 8; ++i) t += red[i];
    float e = __expf(-gammap[0] * t);
    outv[row] = (mode == 1) ? alpha[row] * e : e;
  }
}

__global__ __launch_bounds__(256) void rbf_score(
    const _Float16* __restrict__ xh, const _Float16* __restrict__ Xh,
    const float* __restrict__ sx, const float* __restrict__ w,
    const float* __restrict__ gammap, float* __restrict__ out) {
  __shared__ __align__(16) _Float16 lbuf[2][16 * LDS_STRIDE];
  __shared__ float wsh[DB_PER];

  const int tid  = threadIdx.x;
  const int wave = tid >> 5;     // 0..7
  const int lane = tid & 31;
  const int hi   = lane >> 4;    // half-wave select
  const int ln   = lane & 15;

  const int m0 = blockIdx.x * 256 + wave * 32;   // 32 x-rows per wave (2 M-tiles)
  const int n0 = blockIdx.y * DB_PER;

  const float c2 = 2.0f * gammap[0] * LOG2E;     // k = w[n] * exp2(c2 * dot)

  // stage this block's slice of w into LDS (2 KB)
  for (int i = tid; i < DB_PER; i += 256) wsh[i] = w[n0 + i];

  // ---- Two A tiles (32 x-rows, K=256) resident in registers, WMMA A layout ----
  // lane holds row M = ln (+16 for tile 1); hi selects K sub-ranges
  // {hi*8..+8, 16+hi*8..+8} within each 32-K chunk.
  v16h a0[8], a1[8];
  {
    const _Float16* xr0 = xh + (size_t)(m0 + ln) * FEAT;
    const _Float16* xr1 = xh + (size_t)(m0 + 16 + ln) * FEAT;
    #pragma unroll
    for (int kc = 0; kc < 8; ++kc) {
      v8h lo0 = *(const v8h*)(xr0 + kc * 32 + hi * 8);
      v8h hp0 = *(const v8h*)(xr0 + kc * 32 + 16 + hi * 8);
      a0[kc] = __builtin_shufflevector(lo0, hp0, 0,1,2,3,4,5,6,7,8,9,10,11,12,13,14,15);
      v8h lo1 = *(const v8h*)(xr1 + kc * 32 + hi * 8);
      v8h hp1 = *(const v8h*)(xr1 + kc * 32 + 16 + hi * 8);
      a1[kc] = __builtin_shufflevector(lo1, hp1, 0,1,2,3,4,5,6,7,8,9,10,11,12,13,14,15);
    }
  }

  // C/D layout: VGPR r holds M = r + hi*8
  float sxv0[8], sxv1[8];
  #pragma unroll
  for (int r = 0; r < 8; ++r) {
    sxv0[r] = sx[m0 + hi * 8 + r];
    sxv1[r] = sx[m0 + 16 + hi * 8 + r];
  }

  float score0[8], score1[8];
  #pragma unroll
  for (int r = 0; r < 8; ++r) { score0[r] = 0.f; score1[r] = 0.f; }

  // global -> reg staging of a 16x256 X tile (32 B per thread)
  const int grow = tid >> 4;   // 0..15 (X row within tile)
  const int gchk = tid & 15;   // 16-half chunk along K
  v8h g0, g1;
  {
    const _Float16* p = Xh + (size_t)(n0 + grow) * FEAT + gchk * 16;
    g0 = *(const v8h*)(p);
    g1 = *(const v8h*)(p + 8);
  }
  {
    _Float16* q = lbuf[0] + grow * LDS_STRIDE + gchk * 16;
    *(v8h*)(q)     = g0;
    *(v8h*)(q + 8) = g1;
  }
  __syncthreads();   // covers lbuf[0] and wsh

  const int iters = DB_PER / 16;
  for (int it = 0; it < iters; ++it) {
    const bool more = (it + 1 < iters);
    if (more) {      // prefetch next tile while this one computes
      const _Float16* p = Xh + (size_t)(n0 + (it + 1) * 16 + grow) * FEAT + gchk * 16;
      g0 = *(const v8h*)(p);
      g1 = *(const v8h*)(p + 8);
    }

    // B layout: lane holds column N = ln; halfs K = kc*32 + {hi*8..+8, 16+hi*8..+8}.
    // Each B load feeds TWO WMMAs (independent accumulator chains).
    const _Float16* base = lbuf[it & 1] + ln * LDS_STRIDE;
    v8f acc0 = {}, acc1 = {};
    #pragma unroll
    for (int kc = 0; kc < 8; ++kc) {
      v8h b0 = *(const v8h*)(base + kc * 32 + hi * 8);
      v8h b1 = *(const v8h*)(base + kc * 32 + 16 + hi * 8);
      v16h b = __builtin_shufflevector(b0, b1, 0,1,2,3,4,5,6,7,8,9,10,11,12,13,14,15);
      acc0 = __builtin_amdgcn_wmma_f32_16x16x32_f16(
          false, a0[kc], false, b, (short)0, acc0, false, false);
      acc1 = __builtin_amdgcn_wmma_f32_16x16x32_f16(
          false, a1[kc], false, b, (short)0, acc1, false, false);
    }

    // epilogue: score += w[n] * exp2(c2 * dot)   (one mul + one TRANS per element)
    const float wL = wsh[it * 16 + ln];
    #pragma unroll
    for (int r = 0; r < 8; ++r) {
      score0[r] += wL * FAST_EXP2(c2 * acc0[r]);
      score1[r] += wL * FAST_EXP2(c2 * acc1[r]);
    }

    if (more) {      // store prefetched tile into the other buffer
      _Float16* q = lbuf[(it + 1) & 1] + grow * LDS_STRIDE + gchk * 16;
      *(v8h*)(q)     = g0;
      *(v8h*)(q + 8) = g1;
    }
    __syncthreads(); // RAW for next buffer, WAR for the one just read
  }

  // reduce over the 16 N-lanes within each half-wave
  #pragma unroll
  for (int r = 0; r < 8; ++r) {
    float s0 = score0[r], s1 = score1[r];
    s0 += __shfl_xor(s0, 1, 32);  s1 += __shfl_xor(s1, 1, 32);
    s0 += __shfl_xor(s0, 2, 32);  s1 += __shfl_xor(s1, 2, 32);
    s0 += __shfl_xor(s0, 4, 32);  s1 += __shfl_xor(s1, 4, 32);
    s0 += __shfl_xor(s0, 8, 32);  s1 += __shfl_xor(s1, 8, 32);
    score0[r] = s0;  score1[r] = s1;
  }
  if (ln == 0) {
    #pragma unroll
    for (int r = 0; r < 8; ++r) {
      atomicAdd(&out[m0 + hi * 8 + r],      sxv0[r] * score0[r]);
      atomicAdd(&out[m0 + 16 + hi * 8 + r], sxv1[r] * score1[r]);
    }
  }
}

extern "C" void kernel_launch(void* const* d_in, const int* in_sizes, int n_in,
                              void* d_out, int out_size, void* d_ws, size_t ws_size,
                              hipStream_t stream) {
  (void)in_sizes; (void)n_in; (void)out_size; (void)ws_size;
  const float* x     = (const float*)d_in[0];
  const float* X     = (const float*)d_in[1];
  const float* alpha = (const float*)d_in[2];
  const float* gamma = (const float*)d_in[3];
  float* out = (float*)d_out;

  // workspace: xh(4MB) | Xh(8MB) | sx(32KB) | w(64KB)   (~12.1 MB)
  char* ws = (char*)d_ws;
  _Float16* xh = (_Float16*)ws;
  _Float16* Xh = (_Float16*)(ws + (size_t)BATCH * FEAT * 2);
  float*    sx = (float*)(ws + (size_t)BATCH * FEAT * 2 + (size_t)DB * FEAT * 2);
  float*    w  = (float*)((char*)sx + (size_t)BATCH * 4);

  zero_out<<<(BATCH + 255) / 256, 256, 0, stream>>>(out, BATCH);
  prep_rows<<<BATCH, FEAT, 0, stream>>>(x, xh, sx, nullptr, gamma, 0);
  prep_rows<<<DB,    FEAT, 0, stream>>>(X, Xh, w,  alpha,   gamma, 1);

  dim3 grid(BATCH / 256, DB / DB_PER);   // 32 x 32 workgroups, 8 waves each
  rbf_score<<<grid, 256, 0, stream>>>(xh, Xh, sx, w, gamma, out);
}
